// MultiHeadedSelfAttention_14963666059553
// MI455X (gfx1250) — matrix operand order repaired and checked by reading, not compiled, not run on hardware
//
#include <hip/hip_runtime.h>

typedef __attribute__((ext_vector_type(16))) __bf16 v16bf;
typedef __attribute__((ext_vector_type(8)))  __bf16 v8bf;
typedef __attribute__((ext_vector_type(8)))  float  v8f;
typedef __attribute__((ext_vector_type(4)))  unsigned int v4u;
typedef __attribute__((ext_vector_type(8)))  int v8i;
typedef __attribute__((ext_vector_type(4)))  int v4i;

#define B_  4
#define S_  2048
#define D_  768
#define H_  12
#define DH_ 64
#define M_  (B_*S_)

#if __has_builtin(__builtin_amdgcn_tensor_load_to_lds)
#define HAVE_TDM 1
#else
#define HAVE_TDM 0
#endif

// ---------- helpers ----------
__device__ __forceinline__ unsigned short f2bf(float f) {
  unsigned int u = __float_as_uint(f);
  unsigned int r = u + 0x7FFFu + ((u >> 16) & 1u);
  return (unsigned short)(r >> 16);
}

__device__ __forceinline__ v8f wmma_bf16(v16bf a, v16bf b, v8f c) {
  return __builtin_amdgcn_wmma_f32_16x16x32_bf16(
      false, a, false, b, (short)0, c, false, false);
}

__device__ __forceinline__ v16bf load_afrag(const unsigned short* rowptr) {
  v8bf lo = *(const v8bf*)(rowptr);
  v8bf hi = *(const v8bf*)(rowptr + 16);
  v16bf a;
#pragma unroll
  for (int i = 0; i < 8; ++i) { a[i] = lo[i]; a[8 + i] = hi[i]; }
  return a;
}

__device__ __forceinline__ v16bf load_bfrag(const unsigned short* p) {
  v8bf lo = *(const v8bf*)(p);
  v8bf hi = *(const v8bf*)(p + 16);
  v16bf b;
#pragma unroll
  for (int i = 0; i < 8; ++i) { b[i] = lo[i]; b[8 + i] = hi[i]; }
  return b;
}

// per-lane async copy (ASYNCcnt)
__device__ __forceinline__ void async_b128(unsigned int lds_addr,
                                           const void* sbase,
                                           unsigned int voff) {
  asm volatile("global_load_async_to_lds_b128 %0, %1, %2"
               :
               : "v"(lds_addr), "v"(voff), "s"(sbase)
               : "memory");
}
__device__ __forceinline__ unsigned int lds_off(const void* p) {
  return (unsigned int)(unsigned long long)p;
}

#if HAVE_TDM
// Tensor Data Mover: 2D bf16 tile (rows x cols elements) -> LDS, with hardware
// row padding (pad_amount 16B after every row). D# per CDNA5 ISA ch.8:
//   group0: count=1 | lds_addr | global_addr[56:0] | type=2
//   group1: data_size=2B, pad_enable, pad codes, tensor dims/strides, tile dims
// Toolchain builtin is the 6-arg form: (g0, g1, g2, g3, g4, cpol).
__device__ __forceinline__ void tdm_load_2d(unsigned int lds_addr,
                                            const void* gaddr,
                                            unsigned int cols, unsigned int rows,
                                            unsigned int tensor_d0,
                                            unsigned int tensor_d1,
                                            unsigned long long stride0_elems,
                                            unsigned int pad_interval_code,
                                            unsigned int pad_amount_code) {
  const unsigned long long ga = (unsigned long long)gaddr;
  v4u g0;
  g0[0] = 1u;                                                   // count=1, user desc
  g0[1] = lds_addr;                                             // LDS byte address
  g0[2] = (unsigned int)(ga & 0xFFFFFFFFu);                     // global addr lo
  g0[3] = (unsigned int)((ga >> 32) & 0x1FFFFFFu) | (2u << 30); // addr hi | type=2
  v8i g1;
  const unsigned int w0 = (1u << 16)                    // data_size = 2 bytes
                        | (1u << 20)                    // pad_enable
                        | (pad_interval_code << 22)
                        | (pad_amount_code << 25);
  g1[0] = (int)w0;                                      // wg_mask=0 (not in cluster)
  g1[1] = (int)((tensor_d0 & 0xFFFFu) << 16);           // barrier_addr=0 | d0 lo16
  g1[2] = (int)((tensor_d0 >> 16) | ((tensor_d1 & 0xFFFFu) << 16));
  g1[3] = (int)((tensor_d1 >> 16) | (cols << 16));      // d1 hi16 | tile_dim0
  g1[4] = (int)(rows);                                  // tile_dim1 | tile_dim2=0
  g1[5] = (int)(unsigned int)(stride0_elems & 0xFFFFFFFFu);
  g1[6] = (int)(unsigned int)((stride0_elems >> 32) & 0xFFFFu); // stride0 hi | stride1 lo
  g1[7] = 0;
  const v4i z4 = {0, 0, 0, 0};
  const v8i z8 = {0, 0, 0, 0, 0, 0, 0, 0};
  __builtin_amdgcn_tensor_load_to_lds(g0, g1, z4, z4, z8, 0);
}
#endif

// ---------- stage 0a: x fp32 -> bf16 ----------
__global__ __launch_bounds__(256) void cvt_x_bf16(const float* __restrict__ x,
                                                  unsigned short* __restrict__ xb,
                                                  int n8) {
  int i = blockIdx.x * blockDim.x + threadIdx.x;
  if (i >= n8) return;
  const float4* p = (const float4*)x + (size_t)i * 2;
  float4 a = p[0], b = p[1];
  union { unsigned short us[8]; uint4 u4; } o;
  o.us[0] = f2bf(a.x); o.us[1] = f2bf(a.y); o.us[2] = f2bf(a.z); o.us[3] = f2bf(a.w);
  o.us[4] = f2bf(b.x); o.us[5] = f2bf(b.y); o.us[6] = f2bf(b.z); o.us[7] = f2bf(b.w);
  ((uint4*)xb)[i] = o.u4;
}

// ---------- stage 0b: W [k][n] fp32 -> Wt [n][k] bf16 ----------
__global__ __launch_bounds__(256) void transpose_w_bf16(const float* __restrict__ W,
                                                        unsigned short* __restrict__ Wt) {
  __shared__ float tile[32][33];
  int k0 = blockIdx.x * 32, n0 = blockIdx.y * 32;
  int tx = threadIdx.x & 31, ty = threadIdx.x >> 5;
#pragma unroll
  for (int i = ty; i < 32; i += 8)
    tile[i][tx] = W[(size_t)(k0 + i) * D_ + n0 + tx];
  __syncthreads();
#pragma unroll
  for (int i = ty; i < 32; i += 8)
    Wt[(size_t)(n0 + i) * D_ + k0 + tx] = f2bf(tile[tx][i]);
}

// ---------- stage 1: out = x @ W + b ----------
// x row-block staged via per-lane async (1552B padded rows; stride not TDM-pad
// expressible); W k-tiles (128x32, 80B rows) double-buffered via TDM.
#define XROW 776
#define WROW 40
__global__ __launch_bounds__(128) void qkv_proj(const unsigned short* __restrict__ xb,
                                                const unsigned short* __restrict__ Wt,
                                                const float* __restrict__ bias,
                                                unsigned short* __restrict__ dst,
                                                int transposed) {
  __shared__ unsigned short xs[16 * XROW];
  __shared__ unsigned short wsb[2][128 * WROW];

  const int t = threadIdx.x;
  const int wave = t >> 5, lane = t & 31;
  const int half = lane >> 4, l16 = lane & 15;
  const int m0 = blockIdx.x * 16;
  const int nbase = blockIdx.y * 128;

  const unsigned int xs0 = lds_off(&xs[0]);
  const unsigned int ws0[2] = { lds_off(&wsb[0][0]), lds_off(&wsb[1][0]) };

  // stage x row-block: 12 async chunks per thread
#pragma unroll
  for (int p = 0; p < 12; ++p) {
    const int chunk = t + p * 128;
    const int row = chunk / 96, c = chunk % 96;
    async_b128(xs0 + row * (XROW * 2) + c * 16, xb,
               (unsigned int)((m0 + row) * (D_ * 2) + c * 16));
  }

  const int NK = D_ / 32;   // 24 k-steps
#if HAVE_TDM
  if (wave == 0)            // W tile 0: one TDM op, hardware 16B row pad
    tdm_load_2d(ws0[0], Wt + (size_t)nbase * D_, 32, 128, D_, D_, D_, 3, 3);
#else
#pragma unroll
  for (int p = 0; p < 4; ++p) {
    const int chunk = t + p * 128;
    const int row = chunk >> 2, c = chunk & 3;
    async_b128(ws0[0] + row * (WROW * 2) + c * 16, Wt,
               (unsigned int)((nbase + row) * (D_ * 2) + c * 16));
  }
#endif
  asm volatile("s_wait_asynccnt 0" ::: "memory");   // x row-block resident

  v8f acc[2];
#pragma unroll
  for (int j = 0; j < 2; ++j)
#pragma unroll
    for (int r = 0; r < 8; ++r) acc[j][r] = 0.0f;

  for (int j = 0; j < NK; ++j) {
#if HAVE_TDM
    if (wave == 0) {
      if (j + 1 < NK) {
        tdm_load_2d(ws0[(j + 1) & 1], Wt + (size_t)nbase * D_ + (j + 1) * 32,
                    32, 128, D_, D_, D_, 3, 3);
        __builtin_amdgcn_s_wait_tensorcnt(1);   // tile j done, j+1 in flight
      } else {
        __builtin_amdgcn_s_wait_tensorcnt(0);
      }
    }
#else
    if (j + 1 < NK) {
      const int k0n = (j + 1) * 32;
#pragma unroll
      for (int p = 0; p < 4; ++p) {
        const int chunk = t + p * 128;
        const int row = chunk >> 2, c = chunk & 3;
        async_b128(ws0[(j + 1) & 1] + row * (WROW * 2) + c * 16, Wt,
                   (unsigned int)((nbase + row) * (D_ * 2) + k0n * 2 + c * 16));
      }
      asm volatile("s_wait_asynccnt 4" ::: "memory");
    } else {
      asm volatile("s_wait_asynccnt 0" ::: "memory");
    }
#endif
    __syncthreads();

    const int k0 = j * 32;
    const v16bf a = load_afrag(xs + l16 * XROW + k0 + half * 8);
    const unsigned short* wb = wsb[j & 1];
#pragma unroll
    for (int jj = 0; jj < 2; ++jj) {
      const unsigned short* br = wb + ((wave * 2 + jj) * 16 + l16) * WROW + half * 8;
      acc[jj] = wmma_bf16(a, load_bfrag(br), acc[jj]);
    }
    __syncthreads();
  }

#pragma unroll
  for (int jj = 0; jj < 2; ++jj) {
    const int col = (blockIdx.y * 8 + wave * 2 + jj) * 16 + l16;
    const float bv = bias[col];
    const int h = col >> 6, d = col & 63;
    if (!transposed) {
#pragma unroll
      for (int r = 0; r < 8; ++r) {
        const int m = m0 + half * 8 + r;
        const int b = m >> 11, s = m & 2047;
        dst[(size_t)((b * H_ + h) * S_ + s) * DH_ + d] = f2bf(acc[jj][r] + bv);
      }
    } else {
      const int m = m0 + half * 8;
      const int b = m >> 11, s = m & 2047;
      union { unsigned short us[8]; uint4 u4; } o;
#pragma unroll
      for (int r = 0; r < 8; ++r) o.us[r] = f2bf(acc[jj][r] + bv);
      *(uint4*)(dst + (size_t)((b * H_ + h) * DH_ + d) * S_ + s) = o.u4;
    }
  }
}

// ---------- stage 2: flash attention, K/V tiles via TDM double buffer ----------
#define TROW 72
__global__ __launch_bounds__(128) void attn_flash(const unsigned short* __restrict__ Q,
                                                  const unsigned short* __restrict__ K,
                                                  const unsigned short* __restrict__ Vt,
                                                  const int* __restrict__ mask,
                                                  float* __restrict__ out) {
  __shared__ unsigned short kt[2][64 * TROW];
  __shared__ unsigned short vt[2][64 * TROW];
  __shared__ unsigned short Plds[4][16 * TROW];

  const int t = threadIdx.x;
  const int wave = t >> 5, lane = t & 31;
  const int half = lane >> 4, l16 = lane & 15;
  const int bh = blockIdx.y;
  const int b = bh / H_, h = bh % H_;
  const int q0 = blockIdx.x * 64 + wave * 16;

  const unsigned short* Qb = Q  + (size_t)bh * S_ * DH_;
  const unsigned short* Kb = K  + (size_t)bh * S_ * DH_;
  const unsigned short* Vb = Vt + (size_t)bh * DH_ * S_;
  unsigned short* P = Plds[wave];

  const unsigned int kt0[2] = { lds_off(&kt[0][0]), lds_off(&kt[1][0]) };
  const unsigned int vt0[2] = { lds_off(&vt[0][0]), lds_off(&vt[1][0]) };

  __builtin_prefetch(mask + b * S_, 0, 0);   // global_prefetch_b8

#if HAVE_TDM
  auto issue_kv = [&](int buf, int kv) {
    // K block: 64 rows x 64 elems, row stride DH
    tdm_load_2d(kt0[buf], Kb + (size_t)kv * DH_, 64, 64, DH_, S_, DH_, 4, 3);
    // V^T block: 64 rows x 64 elems, row stride S
    tdm_load_2d(vt0[buf], Vb + kv, 64, 64, S_, DH_, S_, 4, 3);
  };
#else
  auto issue_kv = [&](int buf, int kv) {
#pragma unroll
    for (int p = 0; p < 4; ++p) {
      const int chunk = t + p * 128;
      const int row = chunk >> 3, c = chunk & 7;
      async_b128(kt0[buf] + row * (TROW * 2) + c * 16, Kb,
                 (unsigned int)((kv + row) * (DH_ * 2) + c * 16));
    }
#pragma unroll
    for (int p = 0; p < 4; ++p) {
      const int chunk = t + p * 128;
      const int row = chunk >> 3, c = chunk & 7;
      async_b128(vt0[buf] + row * (TROW * 2) + c * 16, Vb,
                 (unsigned int)(row * (S_ * 2) + kv * 2 + c * 16));
    }
  };
#endif

  v16bf qa[2];
  {
    const unsigned short* qrow = Qb + (size_t)(q0 + l16) * DH_ + half * 8;
#pragma unroll
    for (int tt = 0; tt < 2; ++tt) qa[tt] = load_afrag(qrow + tt * 32);
  }

  float mrow[8], lrow[8];
  v8f o[4];
#pragma unroll
  for (int r = 0; r < 8; ++r) { mrow[r] = -1e30f; lrow[r] = 0.0f; }
#pragma unroll
  for (int n = 0; n < 4; ++n)
#pragma unroll
    for (int r = 0; r < 8; ++r) o[n][r] = 0.0f;

#if HAVE_TDM
  if (wave == 0) issue_kv(0, 0);
#else
  issue_kv(0, 0);
#endif

  const int NB = S_ / 64;
  for (int j = 0; j < NB; ++j) {
#if HAVE_TDM
    if (wave == 0) {
      if (j + 1 < NB) {
        issue_kv((j + 1) & 1, (j + 1) * 64);
        __builtin_amdgcn_s_wait_tensorcnt(2);   // tiles j done, j+1 (2 ops) in flight
      } else {
        __builtin_amdgcn_s_wait_tensorcnt(0);
      }
    }
#else
    if (j + 1 < NB) {
      issue_kv((j + 1) & 1, (j + 1) * 64);
      asm volatile("s_wait_asynccnt 8" ::: "memory");
    } else {
      asm volatile("s_wait_asynccnt 0" ::: "memory");
    }
#endif
    __syncthreads();

    const int kv0 = j * 64;
    const unsigned short* Kt = kt[j & 1];
    const unsigned short* Vl = vt[j & 1];

    // ---- scores = QK^T * 0.125 - 10000*(1-mask) ----
    v8f sc[4];
#pragma unroll
    for (int n = 0; n < 4; ++n) {
      v8f a;
#pragma unroll
      for (int r = 0; r < 8; ++r) a[r] = 0.0f;
      const unsigned short* krow = Kt + (n * 16 + l16) * TROW + half * 8;
#pragma unroll
      for (int tt = 0; tt < 2; ++tt)
        a = wmma_bf16(qa[tt], load_bfrag(krow + tt * 32), a);
      const float madd =
          -10000.0f * (1.0f - (float)mask[b * S_ + kv0 + n * 16 + l16]);
#pragma unroll
      for (int r = 0; r < 8; ++r) a[r] = a[r] * 0.125f + madd;
      sc[n] = a;
    }

    // ---- online softmax ----
    float alpha[8];
#pragma unroll
    for (int r = 0; r < 8; ++r) {
      float mx = fmaxf(fmaxf(sc[0][r], sc[1][r]), fmaxf(sc[2][r], sc[3][r]));
#pragma unroll
      for (int off = 1; off < 16; off <<= 1) mx = fmaxf(mx, __shfl_xor(mx, off, 32));
      const float nm = fmaxf(mrow[r], mx);
      alpha[r] = __expf(mrow[r] - nm);
      mrow[r] = nm;
    }
#pragma unroll
    for (int n = 0; n < 4; ++n)
#pragma unroll
      for (int r = 0; r < 8; ++r) sc[n][r] = __expf(sc[n][r] - mrow[r]);
#pragma unroll
    for (int r = 0; r < 8; ++r) {
      float sm = sc[0][r] + sc[1][r] + sc[2][r] + sc[3][r];
#pragma unroll
      for (int off = 1; off < 16; off <<= 1) sm += __shfl_xor(sm, off, 32);
      lrow[r] = lrow[r] * alpha[r] + sm;
    }
#pragma unroll
    for (int n = 0; n < 4; ++n)
#pragma unroll
      for (int r = 0; r < 8; ++r) o[n][r] *= alpha[r];

    // ---- P -> LDS (bf16) round trip into A-fragment layout ----
#pragma unroll
    for (int n = 0; n < 4; ++n)
#pragma unroll
      for (int r = 0; r < 8; ++r)
        P[(half * 8 + r) * TROW + n * 16 + l16] = f2bf(sc[n][r]);
    asm volatile("s_wait_dscnt 0" ::: "memory");

    v16bf pa[2];
    {
      const unsigned short* prow = P + l16 * TROW + half * 8;
#pragma unroll
      for (int tt = 0; tt < 2; ++tt) pa[tt] = load_afrag(prow + tt * 32);
    }

    // ---- O += P V ----
#pragma unroll
    for (int n = 0; n < 4; ++n) {
      const unsigned short* vrow = Vl + (n * 16 + l16) * TROW + half * 8;
#pragma unroll
      for (int tt = 0; tt < 2; ++tt)
        o[n] = wmma_bf16(pa[tt], load_bfrag(vrow + tt * 32), o[n]);
    }
    __syncthreads();
  }

  float inv[8];
#pragma unroll
  for (int r = 0; r < 8; ++r) inv[r] = 1.0f / lrow[r];
#pragma unroll
  for (int n = 0; n < 4; ++n)
#pragma unroll
    for (int r = 0; r < 8; ++r) {
      const int q = q0 + half * 8 + r;
      out[(size_t)(b * S_ + q) * D_ + h * DH_ + n * 16 + l16] = o[n][r] * inv[r];
    }
}

// ---------- host launcher ----------
extern "C" void kernel_launch(void* const* d_in, const int* in_sizes, int n_in,
                              void* d_out, int out_size, void* d_ws, size_t ws_size,
                              hipStream_t stream) {
  const float* x    = (const float*)d_in[0];
  const int*   mask = (const int*)  d_in[1];
  const float* Wq   = (const float*)d_in[2];
  const float* bq   = (const float*)d_in[3];
  const float* Wk   = (const float*)d_in[4];
  const float* bk   = (const float*)d_in[5];
  const float* Wv   = (const float*)d_in[6];
  const float* bv   = (const float*)d_in[7];
  float* out = (float*)d_out;

  char* ws = (char*)d_ws;
  size_t off = 0;
  auto alloc = [&](size_t bytes) {
    char* p = ws + off;
    off += (bytes + 255) & ~(size_t)255;
    return p;
  };
  unsigned short* xb  = (unsigned short*)alloc((size_t)M_ * D_ * 2);
  unsigned short* Wtq = (unsigned short*)alloc((size_t)D_ * D_ * 2);
  unsigned short* Wtk = (unsigned short*)alloc((size_t)D_ * D_ * 2);
  unsigned short* Wtv = (unsigned short*)alloc((size_t)D_ * D_ * 2);
  unsigned short* Qw  = (unsigned short*)alloc((size_t)M_ * D_ * 2);
  unsigned short* Kw  = (unsigned short*)alloc((size_t)M_ * D_ * 2);
  unsigned short* Vw  = (unsigned short*)alloc((size_t)M_ * D_ * 2);
  (void)in_sizes; (void)n_in; (void)out_size; (void)ws_size;

  const int n8 = (M_ * D_) / 8;
  cvt_x_bf16<<<(n8 + 255) / 256, 256, 0, stream>>>(x, xb, n8);
  transpose_w_bf16<<<dim3(D_ / 32, D_ / 32), 256, 0, stream>>>(Wq, Wtq);
  transpose_w_bf16<<<dim3(D_ / 32, D_ / 32), 256, 0, stream>>>(Wk, Wtk);
  transpose_w_bf16<<<dim3(D_ / 32, D_ / 32), 256, 0, stream>>>(Wv, Wtv);

  dim3 pgrid(M_ / 16, 6);
  qkv_proj<<<pgrid, 128, 0, stream>>>(xb, Wtq, bq, Qw, 0);
  qkv_proj<<<pgrid, 128, 0, stream>>>(xb, Wtk, bk, Kw, 0);
  qkv_proj<<<pgrid, 128, 0, stream>>>(xb, Wtv, bv, Vw, 1);

  attn_flash<<<dim3(S_ / 64, B_ * H_), 128, 0, stream>>>(Qw, Kw, Vw, mask, out);
}
